// DilatedAttention_41223096107201
// MI455X (gfx1250) — matrix operand order repaired and checked by reading, not compiled, not run on hardware
//
#include <hip/hip_runtime.h>
#include <hip/hip_bf16.h>

typedef __attribute__((ext_vector_type(16))) _Float16 v16h;
typedef __attribute__((ext_vector_type(2)))  _Float16 v2h;
typedef __attribute__((ext_vector_type(8)))  float    v8f;
typedef __attribute__((__vector_size__(16))) int      vsi4;   // matches builtin prototype

#define BM 128
#define BN 128
#define BK 32
#define LDSP 40   // halves per LDS row: 32 + 8 pad (80B -> 16B-aligned, bank-rotating)

#define AS_GLOBAL __attribute__((address_space(1)))
#define AS_LDS    __attribute__((address_space(3)))

#if __has_builtin(__builtin_amdgcn_global_load_async_to_lds_b128)
#define ASYNC_COPY 1
#else
#define ASYNC_COPY 0
#endif

__device__ __forceinline__ void wait_async0() {
#if __has_builtin(__builtin_amdgcn_s_wait_asynccnt)
  __builtin_amdgcn_s_wait_asynccnt(0);
#else
  asm volatile("s_wait_asynccnt 0x0" ::: "memory");
#endif
}

__device__ __forceinline__ void async_copy_b128(const _Float16* g, _Float16* l) {
#if ASYNC_COPY
  __builtin_amdgcn_global_load_async_to_lds_b128(
      (AS_GLOBAL vsi4*)g, (AS_LDS vsi4*)l, 0, 0);
#else
  *(uint4*)l = *(const uint4*)g;
#endif
}

union FragH { v16h v; uint4 u[2]; };

// ---------------------------------------------------------------------------
// f32 -> f16 conversion (one-time; inputs are L2-resident, negligible cost)
// ---------------------------------------------------------------------------
__global__ __launch_bounds__(256) void f32_to_f16_kernel(const float* __restrict__ src,
                                                         _Float16* __restrict__ dst, int n) {
  int i = blockIdx.x * blockDim.x + threadIdx.x;
  int stride = gridDim.x * blockDim.x;
  for (; i < n; i += stride) dst[i] = (_Float16)src[i];
}

// ---------------------------------------------------------------------------
// WMMA GEMM: C[M,N] = A[M,K] * B[K,N]  (A,B f16 row-major; this problem:
// 4096x{512,1536}x{768,512,512}, all dims divide the tiling exactly).
// Workgroup: 256 thr = 8 waves; WG tile 128x128; wave tile 64x32 (4x2 frags).
// Double-buffered LDS; A staged via async global->LDS (ASYNCcnt) when the
// toolchain exposes the gfx1250 builtin; B transposed through registers.
// EPI: 0 = store f16, 1 = store f16 + bias, 2 = store f32 + bias
// ---------------------------------------------------------------------------
template <int EPI>
__global__ __launch_bounds__(256) void wmma_gemm_kernel(
    const _Float16* __restrict__ A, const _Float16* __restrict__ B,
    const float* __restrict__ bias, void* __restrict__ Cout,
    int M, int N, int K)
{
  __shared__ _Float16 sA[2][BM * LDSP];   // [m][k] row-major
  __shared__ _Float16 sB[2][BN * LDSP];   // [n][k] (B transposed during staging)

  const int tid  = threadIdx.x;
  const int wave = tid >> 5;
  const int lane = tid & 31;
  const int lrow = lane & 15;
  const int hi   = lane >> 4;          // 0: lanes 0-15, 1: lanes 16-31

  const int bm = blockIdx.y * BM;
  const int bn = blockIdx.x * BN;

  const int waveM = (wave >> 2) * 64;  // 0 or 64
  const int waveN = (wave & 3) * 32;   // 0,32,64,96

  // A staging map: chunk c (8 halves): row=c>>2, kc=(c&3)*8; chunks tid, tid+256
  const int a_row0 = tid >> 2;
  const int a_row1 = a_row0 + 64;
  const int a_kc   = (tid & 3) * 8;
  // B staging map: k-pair p=tid>>4 (k=2p,2p+1), n-group g=tid&15 (n0=g*8)
  const int b_k  = (tid >> 4) * 2;
  const int b_n0 = (tid & 15) * 8;

  const _Float16* aPtr0 = A + (size_t)(bm + a_row0) * K + a_kc;   // + k0
  const _Float16* aPtr1 = A + (size_t)(bm + a_row1) * K + a_kc;   // + k0
  const _Float16* bPtr  = B + (size_t)b_k * N + bn + b_n0;        // + k0*N

  v8f acc[4][2] = {};

  // ---- prologue: stage tile 0 into buffer 0 ----
  async_copy_b128(aPtr0, &sA[0][a_row0 * LDSP + a_kc]);
  async_copy_b128(aPtr1, &sA[0][a_row1 * LDSP + a_kc]);
  {
    uint4 bv0 = *(const uint4*)(bPtr);
    uint4 bv1 = *(const uint4*)(bPtr + N);
    const _Float16* p0 = (const _Float16*)&bv0;
    const _Float16* p1 = (const _Float16*)&bv1;
    #pragma unroll
    for (int i = 0; i < 8; ++i) {
      v2h pr; pr[0] = p0[i]; pr[1] = p1[i];
      *(v2h*)(&sB[0][(b_n0 + i) * LDSP + b_k]) = pr;
    }
  }
#if ASYNC_COPY
  wait_async0();
#endif
  __syncthreads();

  int buf = 0;
  for (int k0 = 0; k0 < K; k0 += BK) {
    const int  nbuf     = buf ^ 1;
    const bool has_next = (k0 + BK) < K;

    // ---- issue next tile's loads before computing this one ----
    uint4 bv0 = {0, 0, 0, 0}, bv1 = {0, 0, 0, 0};
    if (has_next) {
      const int kn = k0 + BK;
      async_copy_b128(aPtr0 + kn, &sA[nbuf][a_row0 * LDSP + a_kc]);
      async_copy_b128(aPtr1 + kn, &sA[nbuf][a_row1 * LDSP + a_kc]);
      bv0 = *(const uint4*)(bPtr + (size_t)kn * N);
      bv1 = *(const uint4*)(bPtr + (size_t)(kn + 1) * N);
    }

    // ---- fragments per ISA 16-bit layouts, from current buffer ----
    FragH afr[4], bfr[2];
    #pragma unroll
    for (int i = 0; i < 4; ++i) {
      // A 16x32: lane row = lane%16; halves 0-7 -> K = hi*8 + 0..7,
      //          halves 8-15 -> K = 16 + hi*8 + 0..7
      const _Float16* base = &sA[buf][(waveM + i * 16 + lrow) * LDSP + hi * 8];
      afr[i].u[0] = *(const uint4*)(base);
      afr[i].u[1] = *(const uint4*)(base + 16);
    }
    #pragma unroll
    for (int j = 0; j < 2; ++j) {
      // B 32x16: lane col = lane%16; halves 0-7 -> K = hi*8 + 0..7,
      //          halves 8-15 -> K = 16 + hi*8 + 0..7
      const _Float16* base = &sB[buf][(waveN + j * 16 + lrow) * LDSP + hi * 8];
      bfr[j].u[0] = *(const uint4*)(base);
      bfr[j].u[1] = *(const uint4*)(base + 16);
    }

    #pragma unroll
    for (int i = 0; i < 4; ++i)
      #pragma unroll
      for (int j = 0; j < 2; ++j)
        acc[i][j] = __builtin_amdgcn_wmma_f32_16x16x32_f16(
            false, afr[i].v, false, bfr[j].v, (short)0, acc[i][j], false, false);

    // ---- finish staging next tile, hand off buffers ----
    if (has_next) {
      const _Float16* p0 = (const _Float16*)&bv0;
      const _Float16* p1 = (const _Float16*)&bv1;
      #pragma unroll
      for (int i = 0; i < 8; ++i) {
        v2h pr; pr[0] = p0[i]; pr[1] = p1[i];
        *(v2h*)(&sB[nbuf][(b_n0 + i) * LDSP + b_k]) = pr;
      }
#if ASYNC_COPY
      wait_async0();
#endif
      __syncthreads();   // uniform: k0/K are uniform across the block
    }
    buf = nbuf;
  }

  // ---- epilogue: D 16x16 f32 layout: lane n = lane%16; vgpr r -> m = r + 8*hi
  #pragma unroll
  for (int i = 0; i < 4; ++i) {
    #pragma unroll
    for (int j = 0; j < 2; ++j) {
      const int col = bn + waveN + j * 16 + lrow;
      float bv = (EPI >= 1) ? bias[col] : 0.0f;
      #pragma unroll
      for (int r = 0; r < 8; ++r) {
        const int row = bm + waveM + i * 16 + hi * 8 + r;
        float v = acc[i][j][r] + bv;
        if (EPI == 2) ((float*)Cout)[(size_t)row * N + col] = v;
        else          ((_Float16*)Cout)[(size_t)row * N + col] = (_Float16)v;
      }
    }
  }
}

// ---------------------------------------------------------------------------
// Dilated attention: per (s,h): 7 keys at t = s + {-24..24 step 8}, softmax,
// weighted V sum. tse row layout: [q(512) | k(512) | v(512)], head stride 64.
// One lane per (s,h) pair; ~58 MFLOP total -> trivially fast.
// ---------------------------------------------------------------------------
__global__ __launch_bounds__(256) void dilated_attn_kernel(
    const _Float16* __restrict__ tse, _Float16* __restrict__ out)
{
  const int gid = blockIdx.x * blockDim.x + threadIdx.x;
  const int h = gid & 7;
  const int s = gid >> 3;
  if (s >= 4096) return;

  const _Float16* qp = tse + (size_t)s * 1536 + h * 64;
  float q[64];
  #pragma unroll
  for (int d = 0; d < 64; ++d) q[d] = (float)qp[d];

  float logits[7];
  #pragma unroll
  for (int w = 0; w < 7; ++w) {
    const int t = s + (w - 3) * 8;
    if (t < 0 || t >= 4096) { logits[w] = -1e30f; continue; }
    const _Float16* kp = tse + (size_t)t * 1536 + 512 + h * 64;
    float dot = 0.0f;
    #pragma unroll
    for (int d = 0; d < 64; ++d) dot += q[d] * (float)kp[d];
    logits[w] = dot * 0.125f;   // 64^-0.5
  }

  float m = logits[0];
  #pragma unroll
  for (int w = 1; w < 7; ++w) m = fmaxf(m, logits[w]);
  float p[7], sum = 0.0f;
  #pragma unroll
  for (int w = 0; w < 7; ++w) { p[w] = __expf(logits[w] - m); sum += p[w]; }
  const float inv = 1.0f / sum;

  float o[64];
  #pragma unroll
  for (int d = 0; d < 64; ++d) o[d] = 0.0f;
  #pragma unroll
  for (int w = 0; w < 7; ++w) {
    const int t = s + (w - 3) * 8;
    if (t < 0 || t >= 4096) continue;
    const _Float16* vp = tse + (size_t)t * 1536 + 1024 + h * 64;
    const float wgt = p[w] * inv;
    #pragma unroll
    for (int d = 0; d < 64; ++d) o[d] += wgt * (float)vp[d];
  }

  _Float16* op = out + (size_t)s * 512 + h * 64;
  #pragma unroll
  for (int d = 0; d < 64; ++d) op[d] = (_Float16)o[d];
}

// ---------------------------------------------------------------------------
extern "C" void kernel_launch(void* const* d_in, const int* in_sizes, int n_in,
                              void* d_out, int out_size, void* d_ws, size_t ws_size,
                              hipStream_t stream) {
  const float* x     = (const float*)d_in[0];   // 4096x768
  const float* W_in  = (const float*)d_in[1];   // 768x512
  const float* b_in  = (const float*)d_in[2];   // 512
  const float* W_tse = (const float*)d_in[3];   // 512x1536
  const float* W_out = (const float*)d_in[4];   // 512x512
  const float* b_out = (const float*)d_in[5];   // 512

  // workspace layout (halves): ~29 MB total
  _Float16* ws     = (_Float16*)d_ws;
  _Float16* x16    = ws;                     // 4096*768  = 3145728
  _Float16* Win16  = x16    + 3145728;       // 393216
  _Float16* Wtse16 = Win16  + 393216;        // 786432
  _Float16* Wout16 = Wtse16 + 786432;        // 262144
  _Float16* h16    = Wout16 + 262144;        // 4096*512  = 2097152
  _Float16* tse16  = h16    + 2097152;       // 4096*1536 = 6291456
  _Float16* att16  = tse16  + 6291456;       // 2097152

  f32_to_f16_kernel<<<512, 256, 0, stream>>>(x,     x16,    4096 * 768);
  f32_to_f16_kernel<<<256, 256, 0, stream>>>(W_in,  Win16,  768 * 512);
  f32_to_f16_kernel<<<256, 256, 0, stream>>>(W_tse, Wtse16, 512 * 1536);
  f32_to_f16_kernel<<<256, 256, 0, stream>>>(W_out, Wout16, 512 * 512);

  // h = x @ W_in + b_in          (4096 x 512, K=768)
  wmma_gemm_kernel<1><<<dim3(512 / BN, 4096 / BM), 256, 0, stream>>>(
      x16, Win16, b_in, h16, 4096, 512, 768);

  // tse = h @ W_tse              (4096 x 1536, K=512)
  wmma_gemm_kernel<0><<<dim3(1536 / BN, 4096 / BM), 256, 0, stream>>>(
      h16, Wtse16, nullptr, tse16, 4096, 1536, 512);

  // windowed dilated attention   (4096*8 lanes)
  dilated_attn_kernel<<<dim3((4096 * 8) / 256), 256, 0, stream>>>(tse16, att16);

  // out = att @ W_out + b_out    (4096 x 512, K=512) -> f32 d_out
  wmma_gemm_kernel<2><<<dim3(512 / BN, 4096 / BM), 256, 0, stream>>>(
      att16, Wout16, b_out, d_out, 4096, 512, 512);
}
